// cost_volume_52604759441834
// MI455X (gfx1250) — matrix operand order repaired and checked by reading, not compiled, not run on hardware
//
#include <hip/hip_runtime.h>
#include <math.h>

// ---------------------------------------------------------------------------
// Problem constants (from reference)
// ---------------------------------------------------------------------------
#define HH   32
#define WW   896
#define BB   2
#define NN   (HH*WW)          // 28672
#define BNT  (BB*NN)          // 57344
#define NSQ  6
#define NSP  4
#define M1T  (BNT*NSQ)        // 344064  (divisible by 48)
#define M2T  (BNT*NSP)        // 229376  (divisible by 64)
#define R1   48               // stage1 rows/block = 8 groups of 6 = 3 row-tiles
#define R2   64               // stage2 rows/block = 16 groups of 4 = 4 row-tiles
#define NEGV (-1.0e10f)
#define DTH  100.0f

// LDS row strides (halves), padded +8 to avoid same-bank row conflicts
#define S1A  200              // stage1 buf0: X1(160)+XZ(32) | layer outs (128)
#define S1B  264              // stage1 buf1: pi_cat(256); scores[0:128)+feat[128:256)
#define S2A  328              // stage2 buf0: pc_cat(320); scores[0:128), pc_pts_g[192:320)
#define S2B  136              // stage2 buf1: X2(32) | layer outs (128)

typedef __attribute__((ext_vector_type(16))) _Float16 v16h;
typedef __attribute__((ext_vector_type(8)))  _Float16 v8h;
typedef __attribute__((ext_vector_type(8)))  float    v8f;

struct LayerP { const _Float16* W; const float* sc; const float* sh; };

// ---------------------------------------------------------------------------
// Weight prep: W(f32, OxK) -> Wp(f16, OxKp zero-padded), scale=g, shift=g*b+beta
// ---------------------------------------------------------------------------
__global__ void prep_weights_k(const float* __restrict__ Wsrc,
                               const float* __restrict__ b,
                               const float* __restrict__ g,
                               const float* __restrict__ beta,
                               int K, int Kp,
                               _Float16* __restrict__ Wp,
                               float* __restrict__ scale,
                               float* __restrict__ shift)
{
    int t = blockIdx.x * blockDim.x + threadIdx.x;
    if (t < 128) { scale[t] = g[t]; shift[t] = g[t] * b[t] + beta[t]; }
    int total = 128 * Kp;
    if (t >= total) return;
    int o = t / Kp, k = t % Kp;
    Wp[t] = (k < K) ? (_Float16)Wsrc[o * K + k] : (_Float16)0.0f;
}

// ---------------------------------------------------------------------------
// select_nearest: 5x15 window, keep 6 nearest (stable on ties, like top_k)
// ---------------------------------------------------------------------------
__global__ void select_q_k(const float* __restrict__ xyz1,
                           const float* __restrict__ xyz2,
                           int* __restrict__ sel, float* __restrict__ mask)
{
    int idx = blockIdx.x * blockDim.x + threadIdx.x;
    if (idx >= BNT) return;
    int b = idx / NN, n = idx % NN;
    int h = n / WW, w = n % WW;
    const float* q = xyz1 + (size_t)idx * 3;
    float qx = q[0], qy = q[1], qz = q[2];
    float dd[NSQ]; int ii[NSQ];
#pragma unroll
    for (int t = 0; t < NSQ; ++t) { dd[t] = 3.4e38f; ii[t] = 0; }
    for (int kh = 0; kh < 5; ++kh) {
        int ch  = h + kh - 2;
        bool hin = (ch >= 0 && ch < HH);
        int chc = ch < 0 ? 0 : (ch > HH - 1 ? HH - 1 : ch);
        for (int kw = 0; kw < 15; ++kw) {
            int cw  = w + kw - 7;
            bool inb = hin && (cw >= 0 && cw < WW);
            int cwc = cw < 0 ? 0 : (cw > WW - 1 ? WW - 1 : cw);
            int cand = chc * WW + cwc;
            const float* c = xyz2 + ((size_t)b * NN + cand) * 3;
            float dx = c[0] - qx, dy = c[1] - qy, dz = c[2] - qz;
            float sq = dx * dx + dy * dy + dz * dz;
            if (inb && sq < DTH && sq < dd[NSQ - 1]) {
                int p = NSQ - 1;
                while (p > 0 && dd[p - 1] > sq) { dd[p] = dd[p - 1]; ii[p] = ii[p - 1]; --p; }
                dd[p] = sq; ii[p] = cand;
            }
        }
    }
#pragma unroll
    for (int t = 0; t < NSQ; ++t) {
        bool v = dd[t] < 3.0e38f;
        sel[idx * NSQ + t]  = v ? ii[t] : 0;
        mask[idx * NSQ + t] = v ? 1.0f : 0.0f;
    }
}

// ---------------------------------------------------------------------------
// select_first: 3x5 window, keep first 4 valid in candidate order
// ---------------------------------------------------------------------------
__global__ void select_p_k(const float* __restrict__ xyz1,
                           int* __restrict__ sel, float* __restrict__ mask)
{
    int idx = blockIdx.x * blockDim.x + threadIdx.x;
    if (idx >= BNT) return;
    int b = idx / NN, n = idx % NN;
    int h = n / WW, w = n % WW;
    const float* q = xyz1 + (size_t)idx * 3;
    float qx = q[0], qy = q[1], qz = q[2];
    int ss[NSP] = {0, 0, 0, 0};
    float mm[NSP] = {0.f, 0.f, 0.f, 0.f};
    int cnt = 0;
    for (int kh = 0; kh < 3; ++kh) {
        int ch  = h + kh - 1;
        bool hin = (ch >= 0 && ch < HH);
        int chc = ch < 0 ? 0 : (ch > HH - 1 ? HH - 1 : ch);
        for (int kw = 0; kw < 5; ++kw) {
            int cw  = w + kw - 2;
            bool inb = hin && (cw >= 0 && cw < WW);
            int cwc = cw < 0 ? 0 : (cw > WW - 1 ? WW - 1 : cw);
            int cand = chc * WW + cwc;
            const float* c = xyz1 + ((size_t)b * NN + cand) * 3;
            float dx = c[0] - qx, dy = c[1] - qy, dz = c[2] - qz;
            float sq = dx * dx + dy * dy + dz * dz;
            if (inb && sq < DTH && cnt < NSP) { ss[cnt] = cand; mm[cnt] = 1.0f; ++cnt; }
        }
    }
#pragma unroll
    for (int t = 0; t < NSP; ++t) {
        sel[idx * NSP + t]  = ss[t];
        mask[idx * NSP + t] = mm[t];
    }
}

// ---------------------------------------------------------------------------
// One WMMA MLP layer over a (NT*16)-row LDS tile: 8 waves, wave w -> cols
// [16w,16w+16). A from LDS, B from global weights, epilogue relu(acc*sc+sh)
// [* rowmask], write to LDS act and/or global.
// CDNA5 16-bit fragment layouts:
//   A: lane<16 holds K[0..7],K[16..23]; lane>=16 holds K[8..15],K[24..31]
//   B: lane<16 holds K[0..7],K[16..23] of col=lane; lane>=16 holds K[8..15],K[24..31] of col=lane-16
// ---------------------------------------------------------------------------
template<int KP, int NT>
__device__ __forceinline__ void mlp_layer(
    const _Float16* actin, int instr,
    const _Float16* __restrict__ Wp,
    const float* __restrict__ scale, const float* __restrict__ shift,
    _Float16* actout, int outstr, int ocol,
    _Float16* __restrict__ gout, long grow0, int gstr,
    const float* rowmask, long mrow0,
    int lane, int wave)
{
    const int col0 = wave * 16;
    const int lrow = lane & 15;
    const int ka = (lane < 16) ? 0 : 8;
    const int kb = (lane < 16) ? 0 : 8;
    const _Float16* wb = Wp + (size_t)(col0 + lrow) * KP + kb;

    v8f acc[NT];
#pragma unroll
    for (int i = 0; i < NT; ++i)
#pragma unroll
        for (int j = 0; j < 8; ++j) acc[i][j] = 0.0f;

#pragma unroll
    for (int k0 = 0; k0 < KP; k0 += 32) {
        v8h blo = *(const v8h*)(wb + k0);
        v8h bhi = *(const v8h*)(wb + k0 + 16);
        v16h bf = __builtin_shufflevector(blo, bhi, 0,1,2,3,4,5,6,7,8,9,10,11,12,13,14,15);
#pragma unroll
        for (int rt = 0; rt < NT; ++rt) {
            const _Float16* ap = actin + (rt * 16 + lrow) * instr + k0 + ka;
            v8h alo = *(const v8h*)ap;
            v8h ahi = *(const v8h*)(ap + 16);
            v16h af = __builtin_shufflevector(alo, ahi, 0,1,2,3,4,5,6,7,8,9,10,11,12,13,14,15);
            acc[rt] = __builtin_amdgcn_wmma_f32_16x16x32_f16(
                false, af, false, bf, (short)0, acc[rt], false, false);
        }
    }

    const int ch = col0 + lrow;
    const float sc = scale[ch], sh = shift[ch];
#pragma unroll
    for (int rt = 0; rt < NT; ++rt) {
#pragma unroll
        for (int r = 0; r < 8; ++r) {
            int row = rt * 16 + ((lane < 16) ? r : r + 8);
            float v = fmaxf(acc[rt][r] * sc + sh, 0.0f);
            if (rowmask) v *= rowmask[mrow0 + row];
            _Float16 hv = (_Float16)v;
            if (actout) actout[row * outstr + ocol + ch] = hv;
            if (gout)   gout[(grow0 + row) * (long)gstr + ch] = hv;
        }
    }
}

// ---------------------------------------------------------------------------
// Fully fused stage 1 (48 rows = 8 groups of 6 per block):
//   LDS build (xyz header + pts1 + pts2*m gathers) -> mlp1.0..2 -> pi_enc ->
//   mlp2.0..1 -> masked softmax over k=6 -> weighted feat sum -> pifeat (f32).
// Only HBM I/O: raw inputs + sel/mask in, pifeat out.
// ---------------------------------------------------------------------------
struct Stage1Args {
    LayerP L[6];                  // mlp1.0, mlp1.1, mlp1.2, pi_enc, mlp2.0, mlp2.1
    const float* xyz1; const float* xyz2;
    const float* pts1; const float* pts2;
    const int*   sel_q; const float* mask_q;
    float* pifeat;                // BNT x 128
};

__global__ __launch_bounds__(256) void stage1_k(Stage1Args A)
{
    __shared__ __align__(16) _Float16 s0[R1 * S1A];
    __shared__ __align__(16) _Float16 s1[R1 * S1B];
    __shared__ int   sj[R1];
    __shared__ float sm[R1];
    const int t = threadIdx.x, lane = t & 31, wave = t >> 5;
    const long rowblk = (long)blockIdx.x * R1;
    const long bn0 = rowblk / NSQ;

    // phase 0: per-row xyz header (10 ch) into X1 cols[0:10) and XZ cols[160:170)
    if (t < R1) {
        long idx = rowblk + t;
        long bn  = idx / NSQ;
        int  b   = (int)(bn / NN);
        float m  = A.mask_q[idx];
        int   j  = A.sel_q[idx];
        sj[t] = j; sm[t] = m;
        const float* p1 = A.xyz1 + (size_t)bn * 3;
        const float* p2 = A.xyz2 + ((size_t)b * NN + j) * 3;
        float pix = p1[0], piy = p1[1], piz = p1[2];
        float qix = p2[0] * m, qiy = p2[1] * m, qiz = p2[2] * m;
        float dx = qix - pix, dy = qiy - piy, dz = qiz - piz;
        float eu = sqrtf(dx * dx + dy * dy + dz * dz + 1e-20f);
        float ten[10] = {pix, piy, piz, qix, qiy, qiz, dx, dy, dz, eu};
        _Float16* r0 = s0 + t * S1A;
#pragma unroll
        for (int c = 0; c < 10; ++c) { r0[c] = (_Float16)ten[c]; r0[160 + c] = (_Float16)ten[c]; }
#pragma unroll
        for (int c = 138; c < 160; ++c) r0[c] = (_Float16)0.0f;
#pragma unroll
        for (int c = 170; c < 192; ++c) r0[c] = (_Float16)0.0f;
    }
    __syncthreads();
    // phase 1: gathers. pts1 -> cols[10:74); pts2*m -> cols[74:138)
    for (int o = t; o < R1 * 16; o += 256) {
        int r = o >> 4, q = o & 15;
        long bn = (rowblk + r) / NSQ;
        const float* f = A.pts1 + (size_t)bn * 64 + q * 4;
        _Float16* d = s0 + r * S1A + 10 + q * 4;
        d[0] = (_Float16)f[0]; d[1] = (_Float16)f[1];
        d[2] = (_Float16)f[2]; d[3] = (_Float16)f[3];
    }
    for (int o = t; o < R1 * 16; o += 256) {
        int r = o >> 4, q = o & 15;
        int  b = (int)((rowblk + r) / ((long)NN * NSQ));
        float m = sm[r]; int j = sj[r];
        const float* f = A.pts2 + ((size_t)b * NN + j) * 64 + q * 4;
        _Float16* d = s0 + r * S1A + 74 + q * 4;
        d[0] = (_Float16)(f[0] * m); d[1] = (_Float16)(f[1] * m);
        d[2] = (_Float16)(f[2] * m); d[3] = (_Float16)(f[3] * m);
    }
    __syncthreads();
    // mlp1.0 : s0(160) -> s1[:,0:128)
    mlp_layer<160,3>(s0, S1A, A.L[0].W, A.L[0].sc, A.L[0].sh, s1, S1B, 0,
                     nullptr, 0, 0, nullptr, 0, lane, wave);
    __syncthreads();
    // mlp1.1 : s1(128) -> s0[:,0:128)  (X1 dead; XZ cols[160:192) preserved)
    mlp_layer<128,3>(s1, S1B, A.L[1].W, A.L[1].sc, A.L[1].sh, s0, S1A, 0,
                     nullptr, 0, 0, nullptr, 0, lane, wave);
    __syncthreads();
    // mlp1.2 : s0(128) -> s1[:,128:256) = feat
    mlp_layer<128,3>(s0, S1A, A.L[2].W, A.L[2].sc, A.L[2].sh, s1, S1B, 128,
                     nullptr, 0, 0, nullptr, 0, lane, wave);
    __syncthreads();
    // pi_enc : s0[:,160:192) (K=32) -> s1[:,0:128)  => s1 = pi_cat(256)
    mlp_layer<32,3>(s0 + 160, S1A, A.L[3].W, A.L[3].sc, A.L[3].sh, s1, S1B, 0,
                    nullptr, 0, 0, nullptr, 0, lane, wave);
    __syncthreads();
    // mlp2.0 : s1(256) -> s0[:,0:128)
    mlp_layer<256,3>(s1, S1B, A.L[4].W, A.L[4].sc, A.L[4].sh, s0, S1A, 0,
                     nullptr, 0, 0, nullptr, 0, lane, wave);
    __syncthreads();
    // mlp2.1 : s0(128) -> s1[:,0:128) = scores (feat at s1[:,128:256) intact)
    mlp_layer<128,3>(s0, S1A, A.L[5].W, A.L[5].sc, A.L[5].sh, s1, S1B, 0,
                     nullptr, 0, 0, nullptr, 0, lane, wave);
    __syncthreads();
    // fused masked softmax over k=6 + weighted feat sum -> pifeat (8 groups x 128)
    for (int o = t; o < 8 * 128; o += 256) {
        int g = o >> 7, c = o & 127;
        float s[NSQ];
        float mx = -3.4e38f;
#pragma unroll
        for (int k = 0; k < NSQ; ++k) {
            int r = g * NSQ + k;
            float v = (sm[r] == 1.0f) ? (float)s1[r * S1B + c] : NEGV;
            s[k] = v; mx = fmaxf(mx, v);
        }
        float sum = 0.0f;
#pragma unroll
        for (int k = 0; k < NSQ; ++k) { s[k] = expf(s[k] - mx); sum += s[k]; }
        float inv = 1.0f / sum;
        float a = 0.0f;
#pragma unroll
        for (int k = 0; k < NSQ; ++k)
            a += s[k] * inv * (float)s1[(g * NSQ + k) * S1B + 128 + c];
        A.pifeat[(bn0 + g) * 128 + c] = a;
    }
}

// ---------------------------------------------------------------------------
// Fully fused stage 2 (64 rows = 16 groups of 4 per block):
//   LDS build (xyz_cat2 header + pts1*m + gather(pifeat)*m) -> pc_enc(*mask) ->
//   mlp2_new.0 (K=320) -> mlp2_new.1 -> masked softmax over k=4 -> d_out (f32).
// ---------------------------------------------------------------------------
struct Stage2Args {
    LayerP L[3];                  // pc_enc, mlp2_new.0, mlp2_new.1
    const float* xyz1; const float* pts1; const float* pifeat;
    const int*   sel_p; const float* mask_p;
    float* out;                   // BNT x 128
};

__global__ __launch_bounds__(256) void stage2_k(Stage2Args A)
{
    __shared__ __align__(16) _Float16 s0[R2 * S2A];
    __shared__ __align__(16) _Float16 s1[R2 * S2B];
    __shared__ int   sj[R2];
    __shared__ float sm[R2];
    const int t = threadIdx.x, lane = t & 31, wave = t >> 5;
    const long rowblk = (long)blockIdx.x * R2;
    const long bn0 = rowblk / NSP;

    // phase 0: per-row xyz_cat2 header -> s1[:,0:10), zeros [10:32)
    if (t < R2) {
        long idx = rowblk + t;
        long bn  = idx / NSP;
        int  b   = (int)(bn / NN);
        float m  = A.mask_p[idx];
        int   j  = A.sel_p[idx];
        sj[t] = j; sm[t] = m;
        const float* pn = A.xyz1 + (size_t)bn * 3;
        const float* pg = A.xyz1 + ((size_t)b * NN + j) * 3;
        float nx = pn[0], ny = pn[1], nz = pn[2];
        float gx = pg[0] * m, gy = pg[1] * m, gz = pg[2] * m;
        float dx = gx - nx, dy = gy - ny, dz = gz - nz;
        float eu = sqrtf(dx * dx + dy * dy + dz * dz + 1e-20f);
        float ten[10] = {nx, ny, nz, gx, gy, gz, dx, dy, dz, eu};
        _Float16* r1 = s1 + t * S2B;
#pragma unroll
        for (int c = 0; c < 10; ++c) r1[c] = (_Float16)ten[c];
#pragma unroll
        for (int c = 10; c < 32; ++c) r1[c] = (_Float16)0.0f;
    }
    __syncthreads();
    // phase 1: pts1*m -> s0 cols[128:192); gather(pifeat)*m -> s0 cols[192:320)
    for (int o = t; o < R2 * 16; o += 256) {
        int r = o >> 4, q = o & 15;
        long bn = (rowblk + r) / NSP;
        float m = sm[r];
        const float* f = A.pts1 + (size_t)bn * 64 + q * 4;
        _Float16* d = s0 + r * S2A + 128 + q * 4;
        d[0] = (_Float16)(f[0] * m); d[1] = (_Float16)(f[1] * m);
        d[2] = (_Float16)(f[2] * m); d[3] = (_Float16)(f[3] * m);
    }
    for (int o = t; o < R2 * 32; o += 256) {
        int r = o >> 5, q = o & 31;
        int  b = (int)((rowblk + r) / ((long)NN * NSP));
        float m = sm[r]; int j = sj[r];
        const float* f = A.pifeat + ((size_t)b * NN + j) * 128 + q * 4;
        _Float16* d = s0 + r * S2A + 192 + q * 4;
        d[0] = (_Float16)(f[0] * m); d[1] = (_Float16)(f[1] * m);
        d[2] = (_Float16)(f[2] * m); d[3] = (_Float16)(f[3] * m);
    }
    __syncthreads();
    // pc_enc : s1(32) -> s0[:,0:128), epilogue * mask (LDS)  => s0 = pc_cat(320)
    mlp_layer<32,4>(s1, S2B, A.L[0].W, A.L[0].sc, A.L[0].sh, s0, S2A, 0,
                    nullptr, 0, 0, sm, 0, lane, wave);
    __syncthreads();
    // mlp2_new.0 : s0(320) -> s1[:,0:128)
    mlp_layer<320,4>(s0, S2A, A.L[1].W, A.L[1].sc, A.L[1].sh, s1, S2B, 0,
                     nullptr, 0, 0, nullptr, 0, lane, wave);
    __syncthreads();
    // mlp2_new.1 : s1(128) -> s0[:,0:128) = scores (pc_pts_g cols[192:320) intact)
    mlp_layer<128,4>(s1, S2B, A.L[2].W, A.L[2].sc, A.L[2].sh, s0, S2A, 0,
                     nullptr, 0, 0, nullptr, 0, lane, wave);
    __syncthreads();
    // fused masked softmax over k=4 + weighted pc_pts_g sum -> out (16 groups x 128)
    for (int o = t; o < 16 * 128; o += 256) {
        int g = o >> 7, c = o & 127;
        float s[NSP];
        float mx = -3.4e38f;
#pragma unroll
        for (int k = 0; k < NSP; ++k) {
            int r = g * NSP + k;
            float v = (sm[r] == 1.0f) ? (float)s0[r * S2A + c] : NEGV;
            s[k] = v; mx = fmaxf(mx, v);
        }
        float sum = 0.0f;
#pragma unroll
        for (int k = 0; k < NSP; ++k) { s[k] = expf(s[k] - mx); sum += s[k]; }
        float inv = 1.0f / sum;
        float a = 0.0f;
#pragma unroll
        for (int k = 0; k < NSP; ++k)
            a += s[k] * inv * (float)s0[(g * NSP + k) * S2A + 192 + c];
        A.out[(bn0 + g) * 128 + c] = a;
    }
}

// ---------------------------------------------------------------------------
// Host launch
// ---------------------------------------------------------------------------
extern "C" void kernel_launch(void* const* d_in, const int* in_sizes, int n_in,
                              void* d_out, int out_size, void* d_ws, size_t ws_size,
                              hipStream_t stream)
{
    (void)in_sizes; (void)n_in; (void)out_size; (void)ws_size;

    const float* xyz1 = (const float*)d_in[0];
    const float* xyz2 = (const float*)d_in[1];
    const float* pts1 = (const float*)d_in[2];
    const float* pts2 = (const float*)d_in[3];
    // Param leaves (jax tree flatten, dict keys sorted):
    // mlp1(3 layers), mlp2(2), mlp2_new(2), pc_enc, pi_enc; each (Wt,b,g,beta)
    const float* const* P = (const float* const*)(d_in + 4);

    char* ws = (char*)d_ws;
    size_t off = 0;
    auto alloc = [&](size_t bytes) -> char* {
        char* p = ws + off;
        off = (off + bytes + 255) & ~(size_t)255;
        return p;
    };

    // layer order: mlp1.0, mlp1.1, mlp1.2, mlp2.0, mlp2.1, m2n.0, m2n.1, pc_enc, pi_enc
    const int Ks [9] = {138, 128, 128, 256, 128, 320, 128, 10, 10};
    const int Kps[9] = {160, 128, 128, 256, 128, 320, 128, 32, 32};
    _Float16* wp[9]; float* sc[9]; float* sh[9];
    for (int i = 0; i < 9; ++i) {
        wp[i] = (_Float16*)alloc((size_t)128 * Kps[i] * 2);
        sc[i] = (float*)alloc(128 * 4);
        sh[i] = (float*)alloc(128 * 4);
    }
    int*   sel_q  = (int*)alloc((size_t)M1T * 4);
    float* mask_q = (float*)alloc((size_t)M1T * 4);
    int*   sel_p  = (int*)alloc((size_t)M2T * 4);
    float* mask_p = (float*)alloc((size_t)M2T * 4);
    float* pifeat = (float*)alloc((size_t)BNT * 128 * 4);

    const float* Wsrc[9] = {P[0], P[4], P[8],  P[12], P[16], P[20], P[24], P[28], P[32]};
    const float* bsrc[9] = {P[1], P[5], P[9],  P[13], P[17], P[21], P[25], P[29], P[33]};
    const float* gsrc[9] = {P[2], P[6], P[10], P[14], P[18], P[22], P[26], P[30], P[34]};
    const float* tsrc[9] = {P[3], P[7], P[11], P[15], P[19], P[23], P[27], P[31], P[35]};
    for (int i = 0; i < 9; ++i) {
        int tot = 128 * Kps[i];
        prep_weights_k<<<(tot + 255) / 256, 256, 0, stream>>>(
            Wsrc[i], bsrc[i], gsrc[i], tsrc[i], Ks[i], Kps[i], wp[i], sc[i], sh[i]);
    }

    // ---- Stage 1 ----
    select_q_k<<<(BNT + 255) / 256, 256, 0, stream>>>(xyz1, xyz2, sel_q, mask_q);
    Stage1Args a1;
    a1.L[0] = {wp[0], sc[0], sh[0]};  // mlp1.0
    a1.L[1] = {wp[1], sc[1], sh[1]};  // mlp1.1
    a1.L[2] = {wp[2], sc[2], sh[2]};  // mlp1.2
    a1.L[3] = {wp[8], sc[8], sh[8]};  // pi_enc
    a1.L[4] = {wp[3], sc[3], sh[3]};  // mlp2.0
    a1.L[5] = {wp[4], sc[4], sh[4]};  // mlp2.1
    a1.xyz1 = xyz1; a1.xyz2 = xyz2; a1.pts1 = pts1; a1.pts2 = pts2;
    a1.sel_q = sel_q; a1.mask_q = mask_q; a1.pifeat = pifeat;
    stage1_k<<<M1T / R1, 256, 0, stream>>>(a1);

    // ---- Stage 2 ----
    select_p_k<<<(BNT + 255) / 256, 256, 0, stream>>>(xyz1, sel_p, mask_p);
    Stage2Args a2;
    a2.L[0] = {wp[7], sc[7], sh[7]};  // pc_enc
    a2.L[1] = {wp[5], sc[5], sh[5]};  // mlp2_new.0
    a2.L[2] = {wp[6], sc[6], sh[6]};  // mlp2_new.1
    a2.xyz1 = xyz1; a2.pts1 = pts1; a2.pifeat = pifeat;
    a2.sel_p = sel_p; a2.mask_p = mask_p; a2.out = (float*)d_out;
    stage2_k<<<M2T / R2, 256, 0, stream>>>(a2);
}
